// SpatialAttention_75866302316553
// MI455X (gfx1250) — matrix operand order repaired and checked by reading, not compiled, not run on hardware
//
#include <hip/hip_runtime.h>
#include <hip/hip_bf16.h>

#define B_  16
#define C_  256
#define HW_ 1024
#define NH_ 8
#define HD_ 32
#define G_  8

typedef __attribute__((ext_vector_type(16))) __bf16 v16bf;
typedef __attribute__((ext_vector_type(8)))  float  v8f;

typedef __attribute__((ext_vector_type(4))) unsigned tdm_g0_t;
typedef __attribute__((ext_vector_type(8))) int      tdm_g1_t;
typedef __attribute__((ext_vector_type(4))) int      tdm_g2_t;
typedef __attribute__((ext_vector_type(8))) int      tdm_g4_t;

static __device__ inline v8f v8f_zero() {
  v8f z;
#pragma unroll
  for (int i = 0; i < 8; ++i) z[i] = 0.0f;
  return z;
}

static __device__ inline v8f wmma_bf16(v16bf a, v16bf b, v8f c) {
  return __builtin_amdgcn_wmma_f32_16x16x32_bf16(
      false, a, false, b, (short)0, c, false, false);
}

static __device__ inline unsigned lds_off(const void* p) {
  // generic LDS pointer: low 32 bits are the LDS byte address (ISA 10.2 aperture rules)
  return (unsigned)(size_t)p;
}

// 1-D TDM copy: nbytes (multiple of 8) from global -> LDS, tracked on TENSORcnt.
static __device__ inline void tdm_load_1d(unsigned lds_addr, const void* gptr, unsigned nbytes) {
  const unsigned long long ga = (unsigned long long)(size_t)gptr;
  const unsigned units = nbytes >> 3;          // data_size=3 (8 bytes)
  tdm_g0_t g0;
  g0[0] = 1u;                                   // count=1, user mode
  g0[1] = lds_addr;                             // lds_addr
  g0[2] = (unsigned)(ga & 0xffffffffu);         // global_addr[31:0]
  g0[3] = (unsigned)((ga >> 32) & 0x01ffffffu)  // global_addr[56:32]
          | (2u << 30);                         // type = 2 (image)
  tdm_g1_t g1;
  const unsigned td0 = 0x40000000u;             // huge tensor_dim0: no OOB clipping
  g1[0] = (int)(3u << 16);                      // wg_mask=0, data_size=3 (8B)
  g1[1] = (int)((td0 & 0xffffu) << 16);         // tensor_dim0[15:0]
  g1[2] = (int)((td0 >> 16) & 0xffffu) | (int)(1u << 16);   // dim0 hi | tensor_dim1=1
  g1[3] = (int)((units & 0xffffu) << 16);       // tile_dim0
  g1[4] = (int)1;                               // tile_dim1=1, tile_dim2=0
  g1[5] = (int)units;                           // tensor_dim0_stride lo
  g1[6] = 0;                                    // stride hi | dim1_stride lo
  g1[7] = 0;
  tdm_g2_t z4; z4[0] = z4[1] = z4[2] = z4[3] = 0;
#if defined(__clang_major__) && (__clang_major__ >= 23)
  tdm_g4_t z8;
#pragma unroll
  for (int i = 0; i < 8; ++i) z8[i] = 0;
  __builtin_amdgcn_tensor_load_to_lds(g0, g1, z4, z4, z8, 0);
#else
  __builtin_amdgcn_tensor_load_to_lds(g0, g1, z4, z4, 0);
#endif
}

// A 16x32 bf16 fragment gather from f32 memory/LDS (contiguous k = 16*(lane/16) + i)
static __device__ inline v16bf load_a_f32(const float* src, int ld, int m0, int k0, int lane) {
  v16bf a;
  const int m  = m0 + (lane & 15);
  const int kb = (lane & 16) ? 8 : 0;
#pragma unroll
  for (int i = 0; i < 16; ++i) {
    const int r = i >> 1, h = i & 1;
    const int k = (kb << 1) + i;
    a[i] = (__bf16)src[(size_t)m * ld + k0 + k];
  }
  return a;
}

// ---- Pack kernels: one fragment (32 lanes x 16 bf16 = 1KB) per wave ----------

// w_qkv [768x256] and w_proj [256x256] -> A-fragments [mt16][kt]
__global__ void __launch_bounds__(128)
pack_w(const float* __restrict__ wqkv, const float* __restrict__ wproj,
       __bf16* __restrict__ Wqf, __bf16* __restrict__ Wpf) {
  const int lane = threadIdx.x & 31;
  const int fid  = blockIdx.x * 4 + (threadIdx.x >> 5);   // 0..511
  const bool isq = fid < 384;
  const int f    = isq ? fid : fid - 384;
  const int mt = f >> 3, kt = f & 7;
  const float* w = isq ? wqkv : wproj;
  __bf16* dst = (isq ? Wqf : Wpf) + (size_t)f * 512 + lane * 16;
  const int m  = mt * 16 + (lane & 15);
  const int kb = (lane & 16) ? 8 : 0;
  v16bf a;
#pragma unroll
  for (int i = 0; i < 16; ++i) {
    const int r = i >> 1, h = i & 1;
    const int k = (kb << 1) + i;
    a[i] = (__bf16)w[(size_t)m * C_ + kt * 32 + k];
  }
  *(v16bf*)dst = a;
}

// x [b][256][1024] f32 -> B-fragments Xf[b][kt=8][st=64]
__global__ void __launch_bounds__(128)
pack_x(const float* __restrict__ x, __bf16* __restrict__ Xf) {
  const int lane = threadIdx.x & 31;
  const int fid  = blockIdx.x * 4 + (threadIdx.x >> 5);   // b*512 + kt*64 + st
  const int b = fid >> 9, r = fid & 511, kt = r >> 6, st = r & 63;
  const float* xb = x + ((size_t)b * C_ + kt * 32) * HW_ + st * 16;
  const int n  = lane & 15;
  const int kb = (lane & 16) ? 16 : 0;
  v16bf v;
#pragma unroll
  for (int i = 0; i < 16; ++i)
    v[i] = (__bf16)xb[(size_t)(kb + i) * HW_ + n];
  *(v16bf*)(Xf + (size_t)fid * 512 + lane * 16) = v;
}

// ---- QKV GEMM: block = 4 waves (M=4x32) x 64 N, sharing TDM-staged B-frags ----
// Outputs consumer-ready fragments: Qf (A-layout, s x d, pre-scaled), Kf (B-layout,
// d x key), Vf (B-layout, key x d).
__global__ void __launch_bounds__(128)
qkv_wmma(const __bf16* __restrict__ Wqf, const __bf16* __restrict__ Xf,
         __bf16* __restrict__ Qf, __bf16* __restrict__ Kf, __bf16* __restrict__ Vf) {
  __shared__ __align__(32) float  sbuf[4][32 * 64];   // 32KB transpose slabs
  __shared__ __align__(32) __bf16 bstage[2][2048];    // 2 x 4KB B-frag stage
  const int lane = threadIdx.x & 31;
  const int wv   = threadIdx.x >> 5;
  const int blk  = blockIdx.x;                 // b*96 + mq*16 + st64
  const int b    = blk / 96;
  const int r    = blk - b * 96;
  const int mq   = r >> 4;
  const int st64 = r & 15;
  const int m32  = mq * 4 + wv;                // 0..23 (32-row tile)
  const int mt16 = m32 * 2;
  const __bf16* xf = Xf + (((size_t)b * 8) * 64 + st64 * 4) * 512;

  if (wv == 0) tdm_load_1d(lds_off(&bstage[0][0]), xf, 4096);

  v8f acc[2][4];
#pragma unroll
  for (int mh = 0; mh < 2; ++mh)
#pragma unroll
    for (int t = 0; t < 4; ++t) acc[mh][t] = v8f_zero();

  for (int kt = 0; kt < 8; ++kt) {
    if (wv == 0) __builtin_amdgcn_s_wait_tensorcnt(0);
    __syncthreads();
    if (wv == 0 && kt + 1 < 8)
      tdm_load_1d(lds_off(&bstage[(kt + 1) & 1][0]),
                  xf + (size_t)(kt + 1) * 64 * 512, 4096);
    const v16bf a0 = *(const v16bf*)(Wqf + ((size_t)mt16 * 8 + kt) * 512 + lane * 16);
    const v16bf a1 = *(const v16bf*)(Wqf + ((size_t)(mt16 + 1) * 8 + kt) * 512 + lane * 16);
    const __bf16* bs = &bstage[kt & 1][0];
#pragma unroll
    for (int t = 0; t < 4; ++t) {
      const v16bf bb = *(const v16bf*)(bs + t * 512 + lane * 16);
      acc[0][t] = wmma_bf16(a0, bb, acc[0][t]);
      acc[1][t] = wmma_bf16(a1, bb, acc[1][t]);
    }
    __syncthreads();
  }

  // transpose 32(d) x 64(s) tile through LDS into consumer fragments
  float* sb = sbuf[wv];
#pragma unroll
  for (int mh = 0; mh < 2; ++mh)
#pragma unroll
    for (int t = 0; t < 4; ++t)
#pragma unroll
      for (int rr = 0; rr < 8; ++rr) {
        const int mr = mh * 16 + rr + ((lane & 16) ? 8 : 0);
        const int nc = t * 16 + (lane & 15);
        sb[mr * 64 + nc] = acc[mh][t][rr];
      }
  asm volatile("s_wait_dscnt 0" ::: "memory");

  const int o0    = m32 * 32;
  const int which = o0 >> 8;              // 0=q 1=k 2=v
  const int h     = (o0 & 255) >> 5;
  const size_t bh = (size_t)b * NH_ + h;
  const int st16  = st64 * 4;

  if (which == 0) {
    const float qs = 0.17677669529663687f;      // hd^-0.5 folded into Q
    const int kb = (lane & 16) ? 8 : 0;
#pragma unroll
    for (int t = 0; t < 4; ++t) {
      v16bf f;
#pragma unroll
      for (int i = 0; i < 16; ++i) {
        const int rI = i >> 1, hI = i & 1;
        const int k = (kb << 1) + i;
        f[i] = (__bf16)(sb[k * 64 + t * 16 + (lane & 15)] * qs);
      }
      *(v16bf*)(Qf + (bh * 64 + st16 + t) * 512 + lane * 16) = f;
    }
  } else if (which == 1) {
    const int kb = (lane & 16) ? 16 : 0;
#pragma unroll
    for (int t = 0; t < 4; ++t) {
      v16bf f;
#pragma unroll
      for (int i = 0; i < 16; ++i)
        f[i] = (__bf16)sb[(kb + i) * 64 + t * 16 + (lane & 15)];
      *(v16bf*)(Kf + (bh * 64 + st16 + t) * 512 + lane * 16) = f;
    }
  } else {
    const int kb = (lane & 16) ? 16 : 0;
#pragma unroll
    for (int jt = 0; jt < 2; ++jt)
#pragma unroll
      for (int dh = 0; dh < 2; ++dh) {
        v16bf f;
#pragma unroll
        for (int i = 0; i < 16; ++i)
          f[i] = (__bf16)sb[(dh * 16 + (lane & 15)) * 64 + jt * 32 + kb + i];
        const int jt32 = st64 * 2 + jt;
        *(v16bf*)(Vf + ((bh * 32 + jt32) * 2 + dh) * 512 + lane * 16) = f;
      }
  }
}

// ---- Flash attention: block = 4 waves of one (b,h); K/V frags TDM-shared ------
__global__ void __launch_bounds__(128)
attn_wmma(const __bf16* __restrict__ Qf, const __bf16* __restrict__ Kf,
          const __bf16* __restrict__ Vf, __bf16* __restrict__ AOf) {
  __shared__ __align__(32) __bf16 kstage[2][1024];   // 2 K-frags / step
  __shared__ __align__(32) __bf16 vstage[2][1024];   // 2 V-frags / step
  __shared__ __align__(32) float  pbuf[4][16 * 32];
  const int lane = threadIdx.x & 31;
  const int wv   = threadIdx.x >> 5;
  const int blk  = blockIdx.x;                 // bh*16 + mq
  const int bh   = blk >> 4;
  const int mt   = (blk & 15) * 4 + wv;        // 16-row query tile, 0..63
  const __bf16* kf = Kf + (size_t)bh * 64 * 512;
  const __bf16* vf = Vf + (size_t)bh * 64 * 512;
  float* pl = pbuf[wv];

  const v16bf qa = *(const v16bf*)(Qf + ((size_t)bh * 64 + mt) * 512 + lane * 16);

  float rmax[8], rsum[8];
  v8f acc0 = v8f_zero(), acc1 = v8f_zero();
#pragma unroll
  for (int i = 0; i < 8; ++i) { rmax[i] = -1e30f; rsum[i] = 0.0f; }

  if (wv == 0) {
    tdm_load_1d(lds_off(&kstage[0][0]), kf, 2048);
    tdm_load_1d(lds_off(&vstage[0][0]), vf, 2048);
  }

  for (int jt = 0; jt < 32; ++jt) {            // 32 keys per step
    if (wv == 0) __builtin_amdgcn_s_wait_tensorcnt(0);
    __syncthreads();
    if (wv == 0 && jt + 1 < 32) {
      tdm_load_1d(lds_off(&kstage[(jt + 1) & 1][0]), kf + (size_t)(jt + 1) * 1024, 2048);
      tdm_load_1d(lds_off(&vstage[(jt + 1) & 1][0]), vf + (size_t)(jt + 1) * 1024, 2048);
    }
    const __bf16* ks = &kstage[jt & 1][0];
    const __bf16* vs = &vstage[jt & 1][0];
    v8f s0 = wmma_bf16(qa, *(const v16bf*)(ks + lane * 16), v8f_zero());
    v8f s1 = wmma_bf16(qa, *(const v16bf*)(ks + 512 + lane * 16), v8f_zero());

#pragma unroll
    for (int rr = 0; rr < 8; ++rr) {
      float mx = fmaxf(s0[rr], s1[rr]);
#pragma unroll
      for (int msk = 1; msk <= 8; msk <<= 1) mx = fmaxf(mx, __shfl_xor(mx, msk, 32));
      const float nm   = fmaxf(rmax[rr], mx);
      const float corr = __expf(rmax[rr] - nm);
      const float p0   = __expf(s0[rr] - nm);
      const float p1   = __expf(s1[rr] - nm);
      float ps = p0 + p1;
#pragma unroll
      for (int msk = 1; msk <= 8; msk <<= 1) ps += __shfl_xor(ps, msk, 32);
      rsum[rr] = rsum[rr] * corr + ps;
      rmax[rr] = nm;
      acc0[rr] *= corr;
      acc1[rr] *= corr;
      const int mr = rr + ((lane & 16) ? 8 : 0);
      const int nc = lane & 15;
      pl[mr * 32 + nc]      = p0;
      pl[mr * 32 + 16 + nc] = p1;
    }
    asm volatile("s_wait_dscnt 0" ::: "memory");
    const v16bf pa = load_a_f32(pl, 32, 0, 0, lane);
    acc0 = wmma_bf16(pa, *(const v16bf*)(vs + lane * 16), acc0);
    acc1 = wmma_bf16(pa, *(const v16bf*)(vs + 512 + lane * 16), acc1);
    __syncthreads();
  }

  // normalize + pack output fragment (B-layout for proj: k=c within head, n=s)
#pragma unroll
  for (int rr = 0; rr < 8; ++rr) {
    const float inv = 1.0f / rsum[rr];
    const int mr = rr + ((lane & 16) ? 8 : 0);
    const int nc = lane & 15;
    pl[mr * 32 + nc]      = acc0[rr] * inv;
    pl[mr * 32 + 16 + nc] = acc1[rr] * inv;
  }
  asm volatile("s_wait_dscnt 0" ::: "memory");
  const int kb = (lane & 16) ? 16 : 0;
  v16bf f;
#pragma unroll
  for (int i = 0; i < 16; ++i)
    f[i] = (__bf16)pl[(lane & 15) * 32 + kb + i];
  *(v16bf*)(AOf + ((size_t)bh * 64 + mt) * 512 + lane * 16) = f;   // AOf[b][kt=h][st]
}

// ---- Projection GEMM: like qkv_wmma, B = AOf fragments, f32 output ------------
__global__ void __launch_bounds__(128)
proj_wmma(const __bf16* __restrict__ Wpf, const __bf16* __restrict__ AOf,
          float* __restrict__ P) {
  __shared__ __align__(32) __bf16 bstage[2][2048];
  const int lane = threadIdx.x & 31;
  const int wv   = threadIdx.x >> 5;
  const int blk  = blockIdx.x;                 // b*32 + msup*16 + st64
  const int b    = blk >> 5;
  const int r    = blk & 31;
  const int msup = r >> 4;
  const int st64 = r & 15;
  const int m32  = msup * 4 + wv;              // 0..7
  const int mt16 = m32 * 2;
  const __bf16* af = AOf + ((size_t)b * 8) * 64 * 512 + (size_t)st64 * 4 * 512;

  if (wv == 0) tdm_load_1d(lds_off(&bstage[0][0]), af, 4096);

  v8f acc[2][4];
#pragma unroll
  for (int mh = 0; mh < 2; ++mh)
#pragma unroll
    for (int t = 0; t < 4; ++t) acc[mh][t] = v8f_zero();

  for (int kt = 0; kt < 8; ++kt) {
    if (wv == 0) __builtin_amdgcn_s_wait_tensorcnt(0);
    __syncthreads();
    if (wv == 0 && kt + 1 < 8)
      tdm_load_1d(lds_off(&bstage[(kt + 1) & 1][0]),
                  af + (size_t)(kt + 1) * 64 * 512, 4096);
    const v16bf a0 = *(const v16bf*)(Wpf + ((size_t)mt16 * 8 + kt) * 512 + lane * 16);
    const v16bf a1 = *(const v16bf*)(Wpf + ((size_t)(mt16 + 1) * 8 + kt) * 512 + lane * 16);
    const __bf16* bs = &bstage[kt & 1][0];
#pragma unroll
    for (int t = 0; t < 4; ++t) {
      const v16bf bb = *(const v16bf*)(bs + t * 512 + lane * 16);
      acc[0][t] = wmma_bf16(a0, bb, acc[0][t]);
      acc[1][t] = wmma_bf16(a1, bb, acc[1][t]);
    }
    __syncthreads();
  }

  float* pb = P + (size_t)b * C_ * HW_;
#pragma unroll
  for (int mh = 0; mh < 2; ++mh)
#pragma unroll
    for (int t = 0; t < 4; ++t)
#pragma unroll
      for (int rr = 0; rr < 8; ++rr) {
        const int o = m32 * 32 + mh * 16 + rr + ((lane & 16) ? 8 : 0);
        const int s = st64 * 64 + t * 16 + (lane & 15);
        pb[(size_t)o * HW_ + s] = acc[mh][t][rr];
      }
}

// ---- GroupNorm stats: one WG per (b,group); region is contiguous --------------
__global__ void __launch_bounds__(256)
gn_stats(const float* __restrict__ P, float* __restrict__ stats) {
  __shared__ float ssum[256], ssq[256];
  const int bg = blockIdx.x;
  const float* base = P + (size_t)bg * (C_ / G_) * HW_;
  float s = 0.0f, q = 0.0f;
  for (int i = threadIdx.x; i < (C_ / G_) * HW_; i += 256) {
    const float v = base[i];
    s += v; q += v * v;
  }
  ssum[threadIdx.x] = s;
  ssq[threadIdx.x]  = q;
  __syncthreads();
  for (int st = 128; st > 0; st >>= 1) {
    if (threadIdx.x < st) {
      ssum[threadIdx.x] += ssum[threadIdx.x + st];
      ssq[threadIdx.x]  += ssq[threadIdx.x + st];
    }
    __syncthreads();
  }
  if (threadIdx.x == 0) {
    const float n    = (float)((C_ / G_) * HW_);
    const float mean = ssum[0] / n;
    const float var  = ssq[0] / n - mean * mean;
    stats[bg * 2]     = mean;
    stats[bg * 2 + 1] = rsqrtf(var + 1e-5f);
  }
}

// ---- GN apply + affine + residual ---------------------------------------------
__global__ void __launch_bounds__(256)
gn_apply(const float* __restrict__ P, const float* __restrict__ x,
         const float* __restrict__ gamma, const float* __restrict__ beta,
         const float* __restrict__ stats, float* __restrict__ out) {
  const size_t i = (size_t)blockIdx.x * 256 + threadIdx.x;
  const int c  = (int)((i >> 10) & 255);
  const int bg = (int)(i >> 15);
  const float mean = stats[bg * 2];
  const float rstd = stats[bg * 2 + 1];
  out[i] = (P[i] - mean) * rstd * gamma[c] + beta[c] + x[i];
}

extern "C" void kernel_launch(void* const* d_in, const int* in_sizes, int n_in,
                              void* d_out, int out_size, void* d_ws, size_t ws_size,
                              hipStream_t stream) {
  (void)in_sizes; (void)n_in; (void)out_size; (void)ws_size;
  const float* x      = (const float*)d_in[0];
  const float* w_qkv  = (const float*)d_in[1];
  const float* w_proj = (const float*)d_in[2];
  const float* gamma  = (const float*)d_in[3];
  const float* beta   = (const float*)d_in[4];
  float* out = (float*)d_out;

  char* ws = (char*)d_ws;
  size_t off = 0;
  __bf16* Wqf = (__bf16*)(ws + off); off += (size_t)384 * 512 * 2;    // 384 KiB
  __bf16* Wpf = (__bf16*)(ws + off); off += (size_t)128 * 512 * 2;    // 128 KiB
  __bf16* Xf  = (__bf16*)(ws + off); off += (size_t)8192 * 512 * 2;   // 8 MiB
  __bf16* Qf  = (__bf16*)(ws + off); off += (size_t)8192 * 512 * 2;
  __bf16* Kf  = (__bf16*)(ws + off); off += (size_t)8192 * 512 * 2;
  __bf16* Vf  = (__bf16*)(ws + off); off += (size_t)8192 * 512 * 2;
  __bf16* AOf = (__bf16*)(ws + off); off += (size_t)8192 * 512 * 2;
  float*  P   = (float*)(ws + off);  off += (size_t)B_ * C_ * HW_ * 4; // 16 MiB
  float*  st  = (float*)(ws + off);

  pack_w   <<<128,  128, 0, stream>>>(w_qkv, w_proj, Wqf, Wpf);
  pack_x   <<<2048, 128, 0, stream>>>(x, Xf);
  qkv_wmma <<<1536, 128, 0, stream>>>(Wqf, Xf, Qf, Kf, Vf);
  attn_wmma<<<2048, 128, 0, stream>>>(Qf, Kf, Vf, AOf);
  proj_wmma<<<512,  128, 0, stream>>>(Wpf, AOf, P);
  gn_stats <<<128,  256, 0, stream>>>(P, st);
  gn_apply <<<16384,256, 0, stream>>>(P, x, gamma, beta, st, out);
}
